// ShiftedWindowAttention_23106924052604
// MI455X (gfx1250) — matrix operand order repaired and checked by reading, not compiled, not run on hardware
//
#include <hip/hip_runtime.h>
#include <hip/hip_bf16.h>
#include <math.h>

typedef __bf16 bf16_t;
typedef __attribute__((ext_vector_type(16))) __bf16 v16bf;
typedef __attribute__((ext_vector_type(8)))  __bf16 v8bf;
typedef __attribute__((ext_vector_type(4)))  __bf16 v4bf;
typedef __attribute__((ext_vector_type(8)))  float  v8f;

#define D_MODEL 256
#define BATCH   8
#define SEQ     8192
#define NTOK    (BATCH * SEQ)   // 65536
#define HEADS   4
#define HDIM    64
#define WIN     5
#define NWIN    1639            // ceil(SEQ / WIN)

#define QKV_ROWS 64             // rows per block in GEMM kernels
#define XS_STRIDE 264           // 256 + 8 bf16 pad (132 dwords -> 4 mod 64 banks)
#define TL_STRIDE 260           // 256 + 4 f32 pad  (260 dwords -> 4 mod 64 banks)

// ---------------------------------------------------------------------------
// WMMA fragment helpers (contiguous k order used by this code, not the ISA table order; 16-bit 16x16x32)
//  A (16x32, MxK): lane l<16 -> row M=l, elems = K[0..15] contiguous (+k0)
//                  lane l>=16 -> row M=l-16, elems = K[16..31]
//  B (32x16, KxN): lane l<16 -> col N=l, elems = K[0..15] contiguous
//                  lane l>=16 -> col N=l-16, elems = K[16..31]
//  C/D (16x16):    lane half h, VGPR r -> (M = 8h + r, N = lane & 15)
// ---------------------------------------------------------------------------
__device__ __forceinline__ v16bf a_frag_ld(const bf16_t* __restrict__ p) {
  const v8bf lo = *reinterpret_cast<const v8bf*>(p);        // K[0..7]
  const v8bf hi = *reinterpret_cast<const v8bf*>(p + 8);    // K[8..15]
  return __builtin_shufflevector(lo, hi, 0, 1, 2, 3, 4, 5, 6, 7,
                                 8, 9, 10, 11, 12, 13, 14, 15);
}

__device__ __forceinline__ v16bf b_frag(const bf16_t* __restrict__ p) {
  return *reinterpret_cast<const v16bf*>(p);   // 32B aligned contiguous
}

#define WMMA_BF16(A, B, C) \
  __builtin_amdgcn_wmma_f32_16x16x32_bf16(false, (A), false, (B), (short)0, (C), false, false)

// ---------------------------------------------------------------------------
// Kernel 1: convert the four 256x256 weight matrices fp32 -> bf16
// ---------------------------------------------------------------------------
__global__ __launch_bounds__(256) void cvt_weights(
    const float* __restrict__ a, const float* __restrict__ b,
    const float* __restrict__ c, const float* __restrict__ d,
    bf16_t* __restrict__ oa, bf16_t* __restrict__ ob,
    bf16_t* __restrict__ oc, bf16_t* __restrict__ od)
{
  const int i = (blockIdx.x * 256 + threadIdx.x) * 4;   // 64 blocks cover 65536
  auto cv = [&](const float* __restrict__ src, bf16_t* __restrict__ dst) {
    const float4 f = *reinterpret_cast<const float4*>(src + i);
    v4bf o;
    o[0] = (bf16_t)f.x; o[1] = (bf16_t)f.y; o[2] = (bf16_t)f.z; o[3] = (bf16_t)f.w;
    *reinterpret_cast<v4bf*>(dst + i) = o;
  };
  cv(a, oa); cv(b, ob); cv(c, oc); cv(d, od);
}

// ---------------------------------------------------------------------------
// Kernel 2: fused QKV projection. Block = 64 rows x (3 x 256) cols, 512 thr.
// x tile converted to bf16 once into LDS; wave w owns cols [16w,16w+16) of
// each of Wq/Wk/Wv across 4 row tiles (12 accumulators, B frags reused 4x).
// out[m][n] = sum_k x[m][k] * W[n][k]  ->  B-matrix column n = W row n.
// ---------------------------------------------------------------------------
__global__ __launch_bounds__(512) void qkv_wmma(
    const float* __restrict__ x,
    const bf16_t* __restrict__ wq, const bf16_t* __restrict__ wk, const bf16_t* __restrict__ wv,
    const float* __restrict__ bq, const float* __restrict__ bk, const float* __restrict__ bv,
    bf16_t* __restrict__ Q, bf16_t* __restrict__ K, bf16_t* __restrict__ V)
{
  __shared__ bf16_t xs[QKV_ROWS][XS_STRIDE];

  const int t    = threadIdx.x;
  const int lane = t & 31;
  const int wave = t >> 5;          // 0..15
  const int nloc = lane & 15;
  const int half = lane >> 4;
  const int rowBase = blockIdx.x * QKV_ROWS;

  // ---- stage x tile (64x256) as bf16 into LDS: 32 elems per thread ----
  {
    const int r  = t >> 3;            // 0..63
    const int c0 = (t & 7) * 32;      // 0,32,..,224
    const float* src = x + (size_t)(rowBase + r) * D_MODEL + c0;
#pragma unroll
    for (int j = 0; j < 32; j += 4) {
      const float4 f = *reinterpret_cast<const float4*>(src + j);
      v4bf o;
      o[0] = (bf16_t)f.x; o[1] = (bf16_t)f.y; o[2] = (bf16_t)f.z; o[3] = (bf16_t)f.w;
      *reinterpret_cast<v4bf*>(&xs[r][c0 + j]) = o;
    }
  }
  __syncthreads();

  const int colBase = wave << 4;    // 16 cols per wave per matrix
  const bf16_t* qp = wq + (size_t)(colBase + nloc) * D_MODEL + half * 16;
  const bf16_t* kp = wk + (size_t)(colBase + nloc) * D_MODEL + half * 16;
  const bf16_t* vp = wv + (size_t)(colBase + nloc) * D_MODEL + half * 16;

  v8f accq[4] = {}, acck[4] = {}, accv[4] = {};
#pragma unroll
  for (int kb = 0; kb < 8; ++kb) {
    const int k0 = kb * 32;
    const v16bf bqf = b_frag(qp + k0);
    const v16bf bkf = b_frag(kp + k0);
    const v16bf bvf = b_frag(vp + k0);
#pragma unroll
    for (int rt = 0; rt < 4; ++rt) {
      const v16bf a = a_frag_ld(&xs[rt * 16 + nloc][k0 + half * 16]);
      accq[rt] = WMMA_BF16(a, bqf, accq[rt]);
      acck[rt] = WMMA_BF16(a, bkf, acck[rt]);
      accv[rt] = WMMA_BF16(a, bvf, accv[rt]);
    }
  }

  const int n = colBase + nloc;
  const float bqv = bq[n], bkv = bk[n], bvv = bv[n];
#pragma unroll
  for (int rt = 0; rt < 4; ++rt) {
#pragma unroll
    for (int r = 0; r < 8; ++r) {
      const size_t m = (size_t)(rowBase + rt * 16 + half * 8 + r);
      Q[m * D_MODEL + n] = (bf16_t)(accq[rt][r] + bqv);
      K[m * D_MODEL + n] = (bf16_t)(acck[rt][r] + bkv);
      V[m * D_MODEL + n] = (bf16_t)(accv[rt][r] + bvv);
    }
  }
}

// ---------------------------------------------------------------------------
// Kernel 3: windowed attention, one wave32 per (batch, head, window).
// Lane owns 2 of the 64 head dims. Zero-padded keys/values participate in
// softmax with score 0, exactly like the reference's pad-then-softmax.
// ---------------------------------------------------------------------------
__global__ __launch_bounds__(256) void win_attn(
    const bf16_t* __restrict__ Q, const bf16_t* __restrict__ K,
    const bf16_t* __restrict__ V, bf16_t* __restrict__ O)
{
  const int lane = threadIdx.x & 31;
  const int wid  = (blockIdx.x << 3) + (threadIdx.x >> 5);
  if (wid >= BATCH * HEADS * NWIN) return;

  const int b   = wid / (HEADS * NWIN);
  const int rem = wid - b * (HEADS * NWIN);
  const int h   = rem / NWIN;
  const int w   = rem - h * NWIN;

  const size_t col = (size_t)h * HDIM + (lane << 1);
  const int s0 = w * WIN;

  float kf0[WIN], kf1[WIN], vf0[WIN], vf1[WIN];
#pragma unroll
  for (int j = 0; j < WIN; ++j) {
    const int s = s0 + j;
    if (s < SEQ) {
      const size_t idx = ((size_t)b * SEQ + s) * D_MODEL + col;
      kf0[j] = (float)K[idx]; kf1[j] = (float)K[idx + 1];
      vf0[j] = (float)V[idx]; vf1[j] = (float)V[idx + 1];
    } else {
      kf0[j] = 0.f; kf1[j] = 0.f; vf0[j] = 0.f; vf1[j] = 0.f;
    }
  }

  for (int i = 0; i < WIN; ++i) {
    const int s = s0 + i;
    if (s >= SEQ) break;                       // wave-uniform
    const size_t idx = ((size_t)b * SEQ + s) * D_MODEL + col;
    const float q0 = (float)Q[idx];
    const float q1 = (float)Q[idx + 1];

    float sc[WIN];
#pragma unroll
    for (int j = 0; j < WIN; ++j) {
      float p = q0 * kf0[j] + q1 * kf1[j];
#pragma unroll
      for (int off = 16; off > 0; off >>= 1) p += __shfl_xor(p, off, 32);
      sc[j] = p * 0.125f;                      // 1/sqrt(64)
    }
    float mx = sc[0];
#pragma unroll
    for (int j = 1; j < WIN; ++j) mx = fmaxf(mx, sc[j]);
    float e[WIN], sum = 0.f;
#pragma unroll
    for (int j = 0; j < WIN; ++j) { e[j] = __expf(sc[j] - mx); sum += e[j]; }
    const float inv = 1.f / sum;
    float o0 = 0.f, o1 = 0.f;
#pragma unroll
    for (int j = 0; j < WIN; ++j) {
      const float p = e[j] * inv;
      o0 += p * vf0[j];
      o1 += p * vf1[j];
    }
    O[idx]     = (bf16_t)tanhf(o0);
    O[idx + 1] = (bf16_t)tanhf(o1);
  }
}

// ---------------------------------------------------------------------------
// Kernel 4: output projection GEMM + bias + residual + LayerNorm, fused.
// Block = 64 rows x 256 cols (512 thr; wave w: Wo col tile w x 4 row tiles),
// staged into a padded LDS slab for the per-row LayerNorm.
// ---------------------------------------------------------------------------
__global__ __launch_bounds__(512) void out_ln_wmma(
    const bf16_t* __restrict__ A, const bf16_t* __restrict__ wo,
    const float* __restrict__ bo, const float* __restrict__ x,
    const float* __restrict__ gamma, const float* __restrict__ beta,
    float* __restrict__ out)
{
  __shared__ float tile[QKV_ROWS][TL_STRIDE];
  __shared__ float partial[2][QKV_ROWS][8];
  __shared__ float stats[2][QKV_ROWS];

  const int t    = threadIdx.x;
  const int lane = t & 31;
  const int wave = t >> 5;
  const int nloc = lane & 15;
  const int half = lane >> 4;
  const int rowBase = blockIdx.x * QKV_ROWS;
  const int colBase = wave << 4;

  const bf16_t* wp = wo + (size_t)(colBase + nloc) * D_MODEL + half * 16;

  v8f acc[4] = {};
#pragma unroll
  for (int kb = 0; kb < 8; ++kb) {
    const int k0 = kb * 32;
    const v16bf bf = b_frag(wp + k0);
#pragma unroll
    for (int rt = 0; rt < 4; ++rt) {
      const v16bf a =
          a_frag_ld(A + (size_t)(rowBase + rt * 16 + nloc) * D_MODEL + k0 + half * 16);
      acc[rt] = WMMA_BF16(a, bf, acc[rt]);
    }
  }

  {
    const int n = colBase + nloc;
    const float bov = bo[n];
#pragma unroll
    for (int rt = 0; rt < 4; ++rt) {
#pragma unroll
      for (int r = 0; r < 8; ++r) {
        const int ml = rt * 16 + half * 8 + r;
        tile[ml][n] = acc[rt][r] + bov + x[(size_t)(rowBase + ml) * D_MODEL + n];
      }
    }
  }
  __syncthreads();

  const int row = t >> 3;        // 0..63
  const int sub = t & 7;         // 0..7
  float s = 0.f, s2 = 0.f;
#pragma unroll
  for (int i = 0; i < 32; ++i) {
    const float v = tile[row][sub + 8 * i];
    s += v; s2 += v * v;
  }
  partial[0][row][sub] = s;
  partial[1][row][sub] = s2;
  __syncthreads();

  if (t < QKV_ROWS) {
    float ts = 0.f, ts2 = 0.f;
#pragma unroll
    for (int i = 0; i < 8; ++i) { ts += partial[0][t][i]; ts2 += partial[1][t][i]; }
    const float mu  = ts * (1.0f / D_MODEL);
    const float var = ts2 * (1.0f / D_MODEL) - mu * mu;
    stats[0][t] = mu;
    stats[1][t] = rsqrtf(var + 1e-5f);
  }
  __syncthreads();

  const float mu   = stats[0][row];
  const float rstd = stats[1][row];
  const size_t g = (size_t)(rowBase + row) * D_MODEL;
#pragma unroll
  for (int i = 0; i < 32; ++i) {
    const int c = sub + 8 * i;
    out[g + c] = (tile[row][c] - mu) * rstd * gamma[c] + beta[c];
  }
}

// ---------------------------------------------------------------------------
// Host launcher
// ---------------------------------------------------------------------------
extern "C" void kernel_launch(void* const* d_in, const int* in_sizes, int n_in,
                              void* d_out, int out_size, void* d_ws, size_t ws_size,
                              hipStream_t stream)
{
  (void)in_sizes; (void)n_in; (void)out_size; (void)ws_size;

  const float* x     = (const float*)d_in[0];
  const float* Wq    = (const float*)d_in[1];
  const float* bq    = (const float*)d_in[2];
  const float* Wk    = (const float*)d_in[3];
  const float* bk    = (const float*)d_in[4];
  const float* Wv    = (const float*)d_in[5];
  const float* bv    = (const float*)d_in[6];
  const float* Wo    = (const float*)d_in[7];
  const float* bo    = (const float*)d_in[8];
  const float* gamma = (const float*)d_in[9];
  const float* beta  = (const float*)d_in[10];
  float* out = (float*)d_out;

  char* ws = (char*)d_ws;
  const size_t mbytes = (size_t)NTOK * D_MODEL * sizeof(bf16_t);   // 33.5 MB
  bf16_t* Qb  = (bf16_t*)(ws + 0 * mbytes);
  bf16_t* Kb  = (bf16_t*)(ws + 1 * mbytes);
  bf16_t* Vb  = (bf16_t*)(ws + 2 * mbytes);
  bf16_t* Ab  = (bf16_t*)(ws + 3 * mbytes);
  bf16_t* wqb = (bf16_t*)(ws + 4 * mbytes);
  bf16_t* wkb = wqb + D_MODEL * D_MODEL;
  bf16_t* wvb = wkb + D_MODEL * D_MODEL;
  bf16_t* wob = wvb + D_MODEL * D_MODEL;

  cvt_weights<<<64, 256, 0, stream>>>(Wq, Wk, Wv, Wo, wqb, wkb, wvb, wob);

  qkv_wmma<<<NTOK / QKV_ROWS, 512, 0, stream>>>(x, wqb, wkb, wvb, bq, bk, bv,
                                                Qb, Kb, Vb);

  const int totalWaves = BATCH * HEADS * NWIN;          // 52448
  win_attn<<<(totalWaves + 7) / 8, 256, 0, stream>>>(Qb, Kb, Vb, Ab);

  out_ln_wmma<<<NTOK / QKV_ROWS, 512, 0, stream>>>(Ab, wob, bo, x, gamma, beta, out);
}
